// VQVAE2_19885698580823
// MI455X (gfx1250) — matrix operand order repaired and checked by reading, not compiled, not run on hardware
//
#include <hip/hip_runtime.h>

// ---------------------------------------------------------------------------
// VQ-VAE-2 forward for MI455X (gfx1250, wave32, WMMA).
// All conv layers run as implicit-GEMM on v_wmma_f32_16x16x32_bf16:
//   M = Cout, N = B*OH*OW, K = Cin*k*k, bf16 A/B, f32 accumulate.
// Bias / pre-ReLU / post-ReLU / residual-add fused into the conv kernel.
// Kernel is templated on <KH, TRANSPOSED> so all filter-index div/mod are
// compile-time; spatial decode uses shifts (all dims are powers of two).
// ---------------------------------------------------------------------------

typedef __attribute__((ext_vector_type(16))) __bf16 v16bf;
typedef __attribute__((ext_vector_type(8)))  __bf16 v8bf;
typedef __attribute__((ext_vector_type(8)))  float  v8f;

#define F_PRE_RELU   1
#define F_POST_RELU  2
#define F_TRANSPOSED 4

template <int KH, bool TR>
__global__ __launch_bounds__(256) void conv_wmma_kernel(
    const float* __restrict__ in, const float* __restrict__ w,
    const float* __restrict__ bias, const float* __restrict__ resid,
    float* __restrict__ out,
    int Cin, int H, int W, int Cout,
    int owShift, int ohwShift, int Npix,
    int stride, int pad, int flags)
{
  __shared__ __attribute__((aligned(16))) __bf16 sA[16 * 32];

  constexpr int KK2 = KH * KH;

  const int tid  = threadIdx.x;
  const int lane = tid & 31;       // wave32
  const int wave = tid >> 5;       // 8 waves per block
  const int col  = lane & 15;      // N column within 16x16 tile
  const int hi   = lane >> 4;      // lane half selects K/M sub-range

  const int K  = Cin * KK2;
  const int m0 = blockIdx.y * 16;                // Cout tile (shared by block)
  const int n0 = (blockIdx.x * 8 + wave) * 16;   // pixel tile (per wave)

  // Decode this lane's output pixel once (all spatial dims are powers of two).
  const int  n      = n0 + col;
  const bool nvalid = n < Npix;
  const int  pix    = nvalid ? n : 0;
  const int  ob     = pix >> ohwShift;
  const int  orem   = pix & ((1 << ohwShift) - 1);
  const int  oy     = orem >> owShift;
  const int  ox     = orem & ((1 << owShift) - 1);

  const bool prelu = (flags & F_PRE_RELU) != 0;
  const float* inb = in + (size_t)ob * Cin * H * W;

  // Hoisted per-lane spatial bases.
  const int by = TR ? (oy + pad) : (oy * stride - pad);
  const int bx = TR ? (ox + pad) : (ox * stride - pad);

  v8f acc = {};

  for (int kk = 0; kk < K; kk += 32) {
    __syncthreads();
    // Stage 16(M) x 32(K) weight tile f32 -> bf16 into LDS (2 elems/thread).
    for (int e = tid; e < 512; e += 256) {
      int ml = e >> 5, kl = e & 31;
      int m = m0 + ml, k = kk + kl;
      float v = 0.f;
      if (m < Cout && k < K) {
        if (TR) {   // torch ConvTranspose2d weight layout (Cin, Cout, kh, kw)
          int ci = k / KK2, r = k % KK2;
          v = w[((size_t)ci * Cout + m) * KK2 + r];
          if (k + 32 < K) {
            int c2 = (k + 32) / KK2, r2 = (k + 32) % KK2;
            __builtin_prefetch(&w[((size_t)c2 * Cout + m) * KK2 + r2], 0, 0);
          }
        } else {    // (Cout, Cin, kh, kw) is row-major M x K
          v = w[(size_t)m * K + k];
          if (k + 32 < K) __builtin_prefetch(&w[(size_t)m * K + k + 32], 0, 0);
        }
      }
      sA[e] = (__bf16)v;
    }
    __syncthreads();

    // A fragment: one contiguous 16-element bf16 run per lane (K = 16*half + j) -> 2x ds_load_b128.
    const __bf16* rowp = &sA[col << 5];
    v8bf alo = *(const v8bf*)(rowp + hi * 16);
    v8bf ahi = *(const v8bf*)(rowp + hi * 16 + 8);
    v16bf a;
#pragma unroll
    for (int j = 0; j < 8; ++j) { a[j] = alo[j]; a[8 + j] = ahi[j]; }

    // B fragment: 32x16 bf16 im2col gather. lane%16 = N, K = 16*half + j.
    v16bf bfr;
#pragma unroll
    for (int j = 0; j < 16; ++j) {
      int k = kk + hi * 16 + j;
      float v = 0.f;
      if (nvalid && k < K) {
        int ci = k / KK2, r = k % KK2;     // compile-time KK2: magic mul / shift
        int ky = r / KH,  kx = r % KH;
        if (!TR) {
          int iy = by + ky, ix = bx + kx;
          if ((unsigned)iy < (unsigned)H && (unsigned)ix < (unsigned)W)
            v = inb[((size_t)ci * H + iy) * W + ix];
        } else {
          int ty = by - ky, tx = bx - kx;
          if (ty >= 0 && tx >= 0 && (ty & (stride - 1)) == 0 &&
              (tx & (stride - 1)) == 0) {
            int iy = ty >> 1, ix = tx >> 1;      // stride == 2 for all convT
            if (iy < H && ix < W) v = inb[((size_t)ci * H + iy) * W + ix];
          }
        }
        if (prelu) v = fmaxf(v, 0.f);
      }
      bfr[j] = (__bf16)v;
    }

    // D = A x B + C, f32 accumulate. EXEC is full here (no wave divergence).
    acc = __builtin_amdgcn_wmma_f32_16x16x32_bf16(false, a, false, bfr,
                                                  (short)0, acc, false, false);
  }

  // Epilogue per D layout (VGPR i -> M = m0 + 8*half + i, N = lane%16).
  const size_t obase = (((size_t)ob * Cout) << ohwShift) + orem;
#pragma unroll
  for (int i = 0; i < 8; ++i) {
    int m = m0 + hi * 8 + i;
    if (nvalid && m < Cout) {
      float v = acc[i] + bias[m];
      size_t oidx = obase + ((size_t)m << ohwShift);
      if (resid) v += resid[oidx];
      if (flags & F_POST_RELU) v = fmaxf(v, 0.f);
      out[oidx] = v;
    }
  }
}

// --------------------------- vector quantization ---------------------------
// One block per spatial position. Forward of straight-through == emb[argmin];
// loss == 1.25 * MSE(z_q, z). Deterministic fixed-order reductions.
__global__ __launch_bounds__(128) void quantize_kernel(
    const float* __restrict__ z, const float* __restrict__ emb,
    float* __restrict__ zq, float* __restrict__ part,
    int HW, int C, int n_e)
{
  __shared__ float zv[64];
  __shared__ float bd[128];
  __shared__ int   bi[128];
  __shared__ float sq[64];

  const int p = blockIdx.x;
  const int b = p / HW, s = p % HW;
  const float* zp = z + (size_t)b * C * HW + s;
  for (int c = threadIdx.x; c < C; c += blockDim.x) zv[c] = zp[(size_t)c * HW];
  __syncthreads();

  float best = 3.4e38f; int bidx = 0;
  for (int e = threadIdx.x; e < n_e; e += blockDim.x) {
    const float* ev = emb + (size_t)e * C;
    float d = 0.f;
#pragma unroll 8
    for (int c = 0; c < 64; ++c) { float t = zv[c] - ev[c]; d = fmaf(t, t, d); }
    if (d < best || (d == best && e < bidx)) { best = d; bidx = e; }
  }
  bd[threadIdx.x] = best; bi[threadIdx.x] = bidx;
  __syncthreads();
  for (int off = 64; off > 0; off >>= 1) {
    if (threadIdx.x < off) {
      if (bd[threadIdx.x + off] < bd[threadIdx.x] ||
          (bd[threadIdx.x + off] == bd[threadIdx.x] &&
           bi[threadIdx.x + off] < bi[threadIdx.x])) {
        bd[threadIdx.x] = bd[threadIdx.x + off];
        bi[threadIdx.x] = bi[threadIdx.x + off];
      }
    }
    __syncthreads();
  }
  const int sel = bi[0];
  if (threadIdx.x < C) {
    float q = emb[(size_t)sel * C + threadIdx.x];
    zq[((size_t)b * C + threadIdx.x) * HW + s] = q;
    float diff = q - zv[threadIdx.x];
    sq[threadIdx.x] = diff * diff;
  }
  __syncthreads();
  if (threadIdx.x == 0) {
    float acc = 0.f;
    for (int c = 0; c < 64; ++c) acc += sq[c];
    part[p] = acc;
  }
}

// ----------------------- bilinear (half-pixel) upsample --------------------
__global__ void upsample_bilinear_kernel(const float* __restrict__ in,
                                         float* __restrict__ out,
                                         int BC, int IH, int IW, int OH, int OW)
{
  int idx = blockIdx.x * blockDim.x + threadIdx.x;
  int total = BC * OH * OW;
  if (idx >= total) return;
  int ox = idx % OW, oy = (idx / OW) % OH, bc = idx / (OW * OH);
  float sx = (ox + 0.5f) * ((float)IW / OW) - 0.5f;
  float sy = (oy + 0.5f) * ((float)IH / OH) - 0.5f;
  int x0 = (int)floorf(sx), y0 = (int)floorf(sy);
  float fx = sx - x0, fy = sy - y0;
  int x1 = min(x0 + 1, IW - 1), y1 = min(y0 + 1, IH - 1);
  x0 = max(x0, 0); y0 = max(y0, 0);
  const float* p = in + (size_t)bc * IH * IW;
  float v00 = p[y0 * IW + x0], v01 = p[y0 * IW + x1];
  float v10 = p[y1 * IW + x0], v11 = p[y1 * IW + x1];
  out[idx] = v00 * (1.f - fy) * (1.f - fx) + v01 * (1.f - fy) * fx +
             v10 * fy * (1.f - fx)         + v11 * fy * fx;
}

// ------------------------- channel-concat copy -----------------------------
__global__ void copy_channels_kernel(const float* __restrict__ src,
                                     float* __restrict__ dst,
                                     int Bn, int Cs, int HW, int Cd, int coff)
{
  int idx = blockIdx.x * blockDim.x + threadIdx.x;
  int total = Bn * Cs * HW;
  if (idx >= total) return;
  int s = idx % HW;
  int c = (idx / HW) % Cs;
  int b = idx / (HW * Cs);
  dst[(((size_t)b * Cd) + coff + c) * HW + s] = src[idx];
}

// ----------------------------- loss finalize -------------------------------
__global__ void loss_finalize_kernel(const float* __restrict__ pb, int nb, float sb,
                                     const float* __restrict__ pt, int nt, float st,
                                     float* __restrict__ dst)
{
  if (threadIdx.x == 0 && blockIdx.x == 0) {
    float ab = 0.f; for (int i = 0; i < nb; ++i) ab += pb[i];
    float at = 0.f; for (int i = 0; i < nt; ++i) at += pt[i];
    dst[0] = ab * sb + at * st;
  }
}

// ------------------------------- host side ---------------------------------
static inline int ilog2(int v) { return 31 - __builtin_clz(v); }

static void conv_launch(hipStream_t st, const float* in, const float* w,
                        const float* b, const float* resid, float* out,
                        int B, int Cin, int H, int W, int Cout,
                        int k, int s, int p, int flags)
{
  int OH, OW;
  if (flags & F_TRANSPOSED) { OH = (H - 1) * s - 2 * p + k; OW = (W - 1) * s - 2 * p + k; }
  else { OH = (H + 2 * p - k) / s + 1; OW = (W + 2 * p - k) / s + 1; }
  int Npix  = B * OH * OW;
  int ntile = (Npix + 15) / 16;
  dim3 grid((ntile + 7) / 8, (Cout + 15) / 16);
  int owS = ilog2(OW), ohwS = ilog2(OH * OW);

  if (flags & F_TRANSPOSED) {
    conv_wmma_kernel<4, true><<<grid, 256, 0, st>>>(
        in, w, b, resid, out, Cin, H, W, Cout, owS, ohwS, Npix, s, p, flags);
  } else if (k == 3) {
    conv_wmma_kernel<3, false><<<grid, 256, 0, st>>>(
        in, w, b, resid, out, Cin, H, W, Cout, owS, ohwS, Npix, s, p, flags);
  } else if (k == 4) {
    conv_wmma_kernel<4, false><<<grid, 256, 0, st>>>(
        in, w, b, resid, out, Cin, H, W, Cout, owS, ohwS, Npix, s, p, flags);
  } else {
    conv_wmma_kernel<1, false><<<grid, 256, 0, st>>>(
        in, w, b, resid, out, Cin, H, W, Cout, owS, ohwS, Npix, s, p, flags);
  }
}

extern "C" void kernel_launch(void* const* d_in, const int* in_sizes, int n_in,
                              void* d_out, int out_size, void* d_ws, size_t ws_size,
                              hipStream_t stream)
{
  (void)in_sizes; (void)n_in; (void)out_size; (void)ws_size;
  #define IN(k) ((const float*)d_in[(k)])

  // Parameter indices: depth-first insertion order of make_params, x first.
  enum {
    HC1W = 1, HC1B, HR1W1, HR1B1, HR1W2, HR1B2, HC2W, HC2B, HR2W1, HR2B1, HR2W2, HR2B2,  // head  1..12
    TC1W, TC1B, TR1W1, TR1B1, TR1W2, TR1B2, TC2W, TC2B, TR2W1, TR2B1, TR2W2, TR2B2,      // enc_top 13..24
    EC1W, EC1B, ER1W1, ER1B1, ER1W2, ER1B2, EC2W, EC2B, ER2W1, ER2B1, ER2W2, ER2B2,      // enc_bottom 25..36
    DBT1W, DBT1B, DBR1W1, DBR1B1, DBR1W2, DBR1B2, DBT2W, DBT2B, DBR2W1, DBR2B1, DBR2W2, DBR2B2, // dec_bottom 37..48
    DTT1W, DTT1B, DTR1W1, DTR1B1, DTR1W2, DTR1B2, DTT2W, DTT2B, DTR2W1, DTR2B1, DTR2W2, DTR2B2, DTOW, DTOB, // dec_top 49..62
    PREQW, PREQB, POSTQW, POSTQB, PREQ2W, PREQ2B, POSTQ2W, POSTQ2B, EMB_B, EMB_T         // 63..72
  };

  const float* X   = IN(0);
  float*       out = (float*)d_out;

  // Arenas. Largest activation = 4x32x256x256 = 8,388,608 floats (= 8Mi).
  // d_out (25.2M floats) doubles as arena A; it is fully rewritten by the
  // final 1x1 conv, so earlier scratch use is safe.
  float* ws   = (float*)d_ws;
  const size_t BIG = (size_t)8 * 1024 * 1024;
  float* A    = out;
  float* Bu   = ws;
  float* Cu   = ws + BIG;
  float* xtop = ws + 2 * BIG;        // 4*64*32*32  = 262144
  float* xbq  = xtop + 262144;       // 4*128*8*8   = 32768
  float* xbqu = xbq + 32768;         // 4*128*32*32 = 524288
  float* cat1 = xbqu + 524288;       // 4*128*32*32 = 524288
  float* cat2 = cat1 + 524288;       // 4*192*32*32 = 786432
  float* pb   = cat2 + 786432;       // 256 bottom-quant partials
  float* pt   = pb + 256;            // 4096 top-quant partials

  const int PP = F_PRE_RELU | F_POST_RELU;

  // ---- head: encb(3 -> 32), 512 -> 128 ----
  conv_launch(stream, X,  IN(HC1W), IN(HC1B), nullptr, A,  4,  3, 512, 512, 32, 4, 2, 1, 0);
  conv_launch(stream, A,  IN(HR1W1), IN(HR1B1), nullptr, Bu, 4, 32, 256, 256, 32, 3, 1, 1, PP);
  conv_launch(stream, Bu, IN(HR1W2), IN(HR1B2), A,       Cu, 4, 32, 256, 256, 32, 1, 1, 0, F_POST_RELU);
  conv_launch(stream, Cu, IN(HC2W), IN(HC2B), nullptr, A,  4, 32, 256, 256, 32, 4, 2, 1, 0);
  conv_launch(stream, A,  IN(HR2W1), IN(HR2B1), nullptr, Bu, 4, 32, 128, 128, 32, 3, 1, 1, PP);
  conv_launch(stream, Bu, IN(HR2W2), IN(HR2B2), A,       Cu, 4, 32, 128, 128, 32, 1, 1, 0, F_POST_RELU);

  // ---- enc_top: encb(32 -> 64), 128 -> 32 ----
  conv_launch(stream, Cu, IN(TC1W), IN(TC1B), nullptr, A,  4, 32, 128, 128, 64, 4, 2, 1, 0);
  conv_launch(stream, A,  IN(TR1W1), IN(TR1B1), nullptr, Bu, 4, 64,  64,  64, 64, 3, 1, 1, PP);
  conv_launch(stream, Bu, IN(TR1W2), IN(TR1B2), A,       Cu, 4, 64,  64,  64, 64, 1, 1, 0, F_POST_RELU);
  conv_launch(stream, Cu, IN(TC2W), IN(TC2B), nullptr, A,  4, 64,  64,  64, 64, 4, 2, 1, 0);
  conv_launch(stream, A,  IN(TR2W1), IN(TR2B1), nullptr, Bu, 4, 64,  32,  32, 64, 3, 1, 1, PP);
  conv_launch(stream, Bu, IN(TR2W2), IN(TR2B2), A,     xtop, 4, 64,  32,  32, 64, 1, 1, 0, F_POST_RELU);

  // ---- enc_bottom: encb(64 -> 128), 32 -> 8 ----
  conv_launch(stream, xtop, IN(EC1W), IN(EC1B), nullptr, A, 4,  64, 32, 32, 128, 4, 2, 1, 0);
  conv_launch(stream, A,  IN(ER1W1), IN(ER1B1), nullptr, Bu, 4, 128, 16, 16, 128, 3, 1, 1, PP);
  conv_launch(stream, Bu, IN(ER1W2), IN(ER1B2), A,       Cu, 4, 128, 16, 16, 128, 1, 1, 0, F_POST_RELU);
  conv_launch(stream, Cu, IN(EC2W), IN(EC2B), nullptr, A,  4, 128, 16, 16, 128, 4, 2, 1, 0);
  conv_launch(stream, A,  IN(ER2W1), IN(ER2B1), nullptr, Bu, 4, 128,  8,  8, 128, 3, 1, 1, PP);
  conv_launch(stream, Bu, IN(ER2W2), IN(ER2B2), A,       Cu, 4, 128,  8,  8, 128, 1, 1, 0, F_POST_RELU);

  // ---- bottom quantize path ----
  conv_launch(stream, Cu, IN(PREQW), IN(PREQB), nullptr, A, 4, 128, 8, 8, 64, 1, 1, 0, 0);
  quantize_kernel<<<256, 128, 0, stream>>>(A, IN(EMB_B), Bu, pb, 64, 64, 256);
  conv_launch(stream, Bu, IN(POSTQW), IN(POSTQB), nullptr, xbq, 4, 64, 8, 8, 128, 1, 1, 0, 0);

  // ---- dec_bottom: 8 -> 32 ----
  conv_launch(stream, xbq, IN(DBT1W), IN(DBT1B), nullptr, A, 4, 128,  8,  8, 128, 4, 2, 1, F_TRANSPOSED);
  conv_launch(stream, A,  IN(DBR1W1), IN(DBR1B1), nullptr, Bu, 4, 128, 16, 16, 128, 3, 1, 1, PP);
  conv_launch(stream, Bu, IN(DBR1W2), IN(DBR1B2), A,       Cu, 4, 128, 16, 16, 128, 1, 1, 0, F_POST_RELU);
  conv_launch(stream, Cu, IN(DBT2W), IN(DBT2B), nullptr, A, 4, 128, 16, 16, 64, 4, 2, 1, F_TRANSPOSED);
  conv_launch(stream, A,  IN(DBR2W1), IN(DBR2B1), nullptr, Bu, 4, 64, 32, 32, 64, 3, 1, 1, PP);
  conv_launch(stream, Bu, IN(DBR2W2), IN(DBR2B2), A,       Cu, 4, 64, 32, 32, 64, 1, 1, 0, F_POST_RELU);

  // ---- concat [xbd, x_top] -> preq2 -> top quantize -> postq2 ----
  copy_channels_kernel<<<(4 * 64 * 1024 + 255) / 256, 256, 0, stream>>>(Cu,   cat1, 4,  64, 1024, 128, 0);
  copy_channels_kernel<<<(4 * 64 * 1024 + 255) / 256, 256, 0, stream>>>(xtop, cat1, 4,  64, 1024, 128, 64);
  conv_launch(stream, cat1, IN(PREQ2W), IN(PREQ2B), nullptr, A, 4, 128, 32, 32, 64, 1, 1, 0, 0);
  quantize_kernel<<<4096, 128, 0, stream>>>(A, IN(EMB_T), Bu, pt, 1024, 64, 2048);
  conv_launch(stream, Bu, IN(POSTQ2W), IN(POSTQ2B), nullptr, Cu, 4, 64, 32, 32, 64, 1, 1, 0, 0);

  // ---- bilinear upsample xbq 8->32, concat [xbq_up, xtq] ----
  upsample_bilinear_kernel<<<(4 * 128 * 1024 + 255) / 256, 256, 0, stream>>>(xbq, xbqu, 4 * 128, 8, 8, 32, 32);
  copy_channels_kernel<<<(4 * 128 * 1024 + 255) / 256, 256, 0, stream>>>(xbqu, cat2, 4, 128, 1024, 192, 0);
  copy_channels_kernel<<<(4 * 64 * 1024 + 255) / 256, 256, 0, stream>>>(Cu,    cat2, 4,  64, 1024, 192, 128);

  // ---- dec_top: 32 -> 128, final 1x1 to 384 channels ----
  conv_launch(stream, cat2, IN(DTT1W), IN(DTT1B), nullptr, A, 4, 192, 32, 32, 64, 4, 2, 1, F_TRANSPOSED);
  conv_launch(stream, A,  IN(DTR1W1), IN(DTR1B1), nullptr, Bu, 4, 64, 64, 64, 64, 3, 1, 1, PP);
  conv_launch(stream, Bu, IN(DTR1W2), IN(DTR1B2), A,       Cu, 4, 64, 64, 64, 64, 1, 1, 0, F_POST_RELU);
  conv_launch(stream, Cu, IN(DTT2W), IN(DTT2B), nullptr, A, 4, 64, 64, 64, 64, 4, 2, 1, F_TRANSPOSED);
  conv_launch(stream, A,  IN(DTR2W1), IN(DTR2B1), nullptr, Bu, 4, 64, 128, 128, 64, 3, 1, 1, PP);
  conv_launch(stream, Bu, IN(DTR2W2), IN(DTR2B2), A,       Cu, 4, 64, 128, 128, 64, 1, 1, 0, 0); // no outer ReLU
  conv_launch(stream, Cu, IN(DTOW), IN(DTOB), nullptr, out, 4, 64, 128, 128, 384, 1, 1, 0, F_POST_RELU);

  // ---- loss = 1.25*MSE_bottom + 1.25*MSE_top, deterministic sum ----
  loss_finalize_kernel<<<1, 32, 0, stream>>>(pb, 256,  1.25f / (256.f * 64.f),
                                             pt, 4096, 1.25f / (4096.f * 64.f),
                                             out + (size_t)4 * 384 * 128 * 128);
  #undef IN
}
